// MultiScaleTemporalAttention_31748398252738
// MI455X (gfx1250) — compile-verified, matrix operand order checked by reading, not run on hardware
//
#include <hip/hip_runtime.h>

// ---------------------------------------------------------------------------
// Types for CDNA5 WMMA
// ---------------------------------------------------------------------------
typedef __attribute__((ext_vector_type(16))) __bf16 v16bf;
typedef __attribute__((ext_vector_type(8)))  float  v8f;

union FragBf16 {
    v16bf v;
    uint4 q[2];   // 2 x 16 bytes = 32 bytes = 16 bf16
};

// ---------------------------------------------------------------------------
// Helpers: f32 -> bf16 (round-to-nearest-even, NaN not special-cased)
// ---------------------------------------------------------------------------
__device__ __forceinline__ unsigned short f2bf(float f) {
    unsigned int u = __float_as_uint(f);
    unsigned int r = u + 0x7FFFu + ((u >> 16) & 1u);
    return (unsigned short)(r >> 16);
}

// ---------------------------------------------------------------------------
// Kernel: f32 -> bf16 elementwise conversion
// ---------------------------------------------------------------------------
__global__ void k_f32_to_bf16(const float* __restrict__ in,
                              unsigned short* __restrict__ out, int n) {
    int i = blockIdx.x * blockDim.x + threadIdx.x;
    if (i < n) out[i] = f2bf(in[i]);
}

// ---------------------------------------------------------------------------
// Kernel: bf16 GEMM  C[M,N] = A[M,K] * B[N,K]^T + bias[N]
// Columns < qcols get (acc+bias) * qscale (folds attention 1/sqrt(dh) into Q).
// One wave computes a 32x64 tile: per 32-wide K-step, 2 A-frags are loaded,
// then each of 4 B-frags is loaded once and feeds two WMMAs (one per A row
// tile). Live VGPRs ~= 64 acc + 16 A + 8 B, so no spilling; B-loads for tile
// t+1 issue while the two WMMAs for tile t execute.
// ---------------------------------------------------------------------------
__global__ void __launch_bounds__(256)
k_gemm_bf16(const unsigned short* __restrict__ A,
            const unsigned short* __restrict__ Bm,
            const float* __restrict__ bias,
            float* __restrict__ C,
            int M, int N, int K, int qcols, float qscale) {
    const int lane = threadIdx.x & 31;
    const int wid  = threadIdx.x >> 5;
    const int wg   = blockIdx.x * (blockDim.x >> 5) + wid;

    const int tilesN = N >> 6;                 // 64-wide column tiles
    const int tilesM = (M + 31) >> 5;          // 32-tall row tiles
    if (wg >= tilesM * tilesN) return;

    const int tm = wg / tilesN;
    const int tn = wg % tilesN;

    const int mrow = lane & 15;
    const int hi   = (lane >> 4) * 8;          // K offset for hi-half lanes

    // A fragment source rows (clamped: padded rows read row M-1, never stored)
    const unsigned short* arow0;
    const unsigned short* arow1;
    {
        int r0 = tm * 32 + mrow;      if (r0 > M - 1) r0 = M - 1;
        int r1 = tm * 32 + 16 + mrow; if (r1 > M - 1) r1 = M - 1;
        arow0 = A + (size_t)r0 * K + hi;
        arow1 = A + (size_t)r1 * K + hi;
    }

    const unsigned short* brow[4];
#pragma unroll
    for (int t = 0; t < 4; ++t)
        brow[t] = Bm + (size_t)(tn * 64 + t * 16 + mrow) * K + hi;

    v8f acc[2][4];
#pragma unroll
    for (int mt = 0; mt < 2; ++mt)
#pragma unroll
        for (int t = 0; t < 4; ++t) acc[mt][t] = v8f{};

    for (int k0 = 0; k0 < K; k0 += 32) {
        FragBf16 a0, a1;
        a0.q[0] = *(const uint4*)(arow0 + k0);
        a0.q[1] = *(const uint4*)(arow0 + k0 + 16);
        a1.q[0] = *(const uint4*)(arow1 + k0);
        a1.q[1] = *(const uint4*)(arow1 + k0 + 16);
#pragma unroll
        for (int t = 0; t < 4; ++t) {
            FragBf16 b;
            b.q[0] = *(const uint4*)(brow[t] + k0);
            b.q[1] = *(const uint4*)(brow[t] + k0 + 16);
            acc[0][t] = __builtin_amdgcn_wmma_f32_16x16x32_bf16(
                false, a0.v, false, b.v, (short)0, acc[0][t], false, false);
            acc[1][t] = __builtin_amdgcn_wmma_f32_16x16x32_bf16(
                false, a1.v, false, b.v, (short)0, acc[1][t], false, false);
        }
    }

    // Store: element r of v8f -> row tm*32 + mt*16 + hi + r,
    //        col tn*64 + t*16 + (lane&15)
#pragma unroll
    for (int t = 0; t < 4; ++t) {
        const int col = tn * 64 + t * 16 + mrow;
        const float bv = bias[col];
        const float f  = (col < qcols) ? qscale : 1.0f;
#pragma unroll
        for (int mt = 0; mt < 2; ++mt) {
#pragma unroll
            for (int r = 0; r < 8; ++r) {
                const int m = tm * 32 + mt * 16 + hi + r;
                if (m < M) C[(size_t)m * N + col] = (acc[mt][t][r] + bv) * f;
            }
        }
    }
}

// ---------------------------------------------------------------------------
// Kernel: windowed attention, mean over window queries.
//   grid.x = Nw (window positions), block = 256 threads = 8 waves (wave = b).
//   obar[b*Nw+n, h*64+d] = sum_k cbar[k] * V[b, n+k, h*64+d]   (bf16 out)
//   cbar[k] = (1/w) sum_q softmax_k(Q[n+q].K[n+k])   (Q pre-scaled by 1/8)
// ---------------------------------------------------------------------------
#define MAXW 10
__global__ void k_window_attn(const float* __restrict__ QKV,   // [B*T, 1536]
                              unsigned short* __restrict__ obar,
                              int w, int Nw) {
    __shared__ float sQ[8][MAXW * 64];
    __shared__ float sK[8][MAXW * 64];
    __shared__ float sS[8][MAXW * MAXW + 28];
    __shared__ float sC[8][16];

    const int lane = threadIdx.x & 31;
    const int b    = threadIdx.x >> 5;   // wave id == batch index
    const int n    = blockIdx.x;
    const int T    = 512;

    for (int h = 0; h < 8; ++h) {
        // Stage Q,K head slab for rows n..n+w-1 into LDS
        for (int i = lane; i < w * 64; i += 32) {
            const int t = i >> 6, d = i & 63;
            const size_t base = ((size_t)(b * T + n + t)) * 1536 + h * 64 + d;
            sQ[b][i] = QKV[base];          // Q part (already * 0.125)
            sK[b][i] = QKV[base + 512];    // K part
        }
        __syncthreads();

        // Scores: lane p covers pairs p, p+32, ... (lq = p/w, lk = p%w)
        for (int p = lane; p < w * w; p += 32) {
            const int lq = p / w, lk = p % w;
            const float4* q4 = (const float4*)&sQ[b][lq * 64];
            const float4* k4 = (const float4*)&sK[b][lk * 64];
            float acc = 0.0f;
#pragma unroll
            for (int i = 0; i < 16; ++i) {
                const float4 qa = q4[i], kb = k4[i];
                acc += qa.x * kb.x + qa.y * kb.y + qa.z * kb.z + qa.w * kb.w;
            }
            sS[b][p] = acc;
        }
        __syncthreads();

        // Row softmax (lane = query row)
        if (lane < w) {
            float* row = &sS[b][lane * w];
            float mx = row[0];
            for (int k = 1; k < w; ++k) mx = fmaxf(mx, row[k]);
            float ssum = 0.0f;
            for (int k = 0; k < w; ++k) { float e = __expf(row[k] - mx); row[k] = e; ssum += e; }
            const float inv = 1.0f / ssum;
            for (int k = 0; k < w; ++k) row[k] *= inv;
        }
        __syncthreads();

        // Column average -> cbar (lane = key column)
        if (lane < w) {
            float c = 0.0f;
            for (int q = 0; q < w; ++q) c += sS[b][q * w + lane];
            sC[b][lane] = c / (float)w;
        }
        __syncthreads();

        // Output: 64 dims over 32 lanes (2 each), weighted sum of V rows
        const int d2 = lane * 2;
        float ax = 0.0f, ay = 0.0f;
        for (int k = 0; k < w; ++k) {
            const float c = sC[b][k];
            const float2 vv = *(const float2*)(QKV +
                ((size_t)(b * T + n + k)) * 1536 + 1024 + h * 64 + d2);
            ax += c * vv.x;
            ay += c * vv.y;
        }
        const unsigned int pair = (unsigned int)f2bf(ax) |
                                  ((unsigned int)f2bf(ay) << 16);
        *(unsigned int*)(obar + ((size_t)(b * Nw + n)) * 512 + h * 64 + d2) = pair;
        __syncthreads();   // before next head reuses LDS
    }
}

// ---------------------------------------------------------------------------
// Kernel: linear interpolation Nw -> T along time, accumulate 0.25 * value
// ---------------------------------------------------------------------------
__global__ void k_interp_acc(const float* __restrict__ Y,  // [B, Nw, D]
                             float* __restrict__ out,      // [B, T, D]
                             int Nw, int first) {
    const int T = 512, D = 512;
    const int idx = blockIdx.x * blockDim.x + threadIdx.x;
    if (idx >= 8 * T * D) return;
    const int d = idx & (D - 1);
    const int t = (idx / D) & (T - 1);
    const int b = idx / (T * D);

    float src = ((float)t + 0.5f) * ((float)Nw / (float)T) - 0.5f;
    src = fminf(fmaxf(src, 0.0f), (float)(Nw - 1));
    const int i0 = (int)floorf(src);
    const int i1 = (i0 + 1 < Nw) ? i0 + 1 : Nw - 1;
    const float fr = src - (float)i0;

    const float v = Y[((size_t)b * Nw + i0) * D + d] * (1.0f - fr) +
                    Y[((size_t)b * Nw + i1) * D + d] * fr;
    const float contrib = 0.25f * v;
    out[idx] = first ? contrib : out[idx] + contrib;
}

// ---------------------------------------------------------------------------
// Launch: x(8,512,512) Wi(4,1536,512) bi(4,1536) Wo(4,512,512) bo(4,512)
// ---------------------------------------------------------------------------
extern "C" void kernel_launch(void* const* d_in, const int* in_sizes, int n_in,
                              void* d_out, int out_size, void* d_ws, size_t ws_size,
                              hipStream_t stream) {
    const float* x  = (const float*)d_in[0];
    const float* Wi = (const float*)d_in[1];
    const float* bi = (const float*)d_in[2];
    const float* Wo = (const float*)d_in[3];
    const float* bo = (const float*)d_in[4];
    float* out = (float*)d_out;

    char* ws = (char*)d_ws;
    unsigned short* xh   = (unsigned short*)(ws);                         //  4 MB
    unsigned short* Wih  = (unsigned short*)(ws + (4u  << 20));           //  1.5 MB
    unsigned short* Woh  = (unsigned short*)(ws + (6u  << 20));           //  0.5 MB
    float*          QKV  = (float*)         (ws + (7u  << 20));           // 24 MB
    unsigned short* obar = (unsigned short*)(ws + (31u << 20));           //  4 MB
    float*          Y    = (float*)         (ws + (35u << 20));           //  8 MB

    const int B = 8, T = 512, D = 512;
    const int scales[4] = {2, 4, 8, 10};

    // x -> bf16 once
    {
        const int n = B * T * D;
        k_f32_to_bf16<<<(n + 255) / 256, 256, 0, stream>>>(x, xh, n);
    }

    for (int s = 0; s < 4; ++s) {
        const int w  = scales[s];
        const int Nw = T - w + 1;

        // Wi[s] -> bf16
        {
            const int n = 3 * D * D;
            k_f32_to_bf16<<<(n + 255) / 256, 256, 0, stream>>>(
                Wi + (size_t)s * n, Wih, n);
        }

        // QKV = x @ Wi^T + bi  (Q columns scaled by 1/sqrt(64))
        {
            const int M = B * T, N = 3 * D, K = D;
            const int tiles = ((M + 31) / 32) * (N / 64);
            k_gemm_bf16<<<(tiles + 7) / 8, 256, 0, stream>>>(
                xh, Wih, bi + (size_t)s * N, QKV, M, N, K, /*qcols=*/D, 0.125f);
        }

        // windowed attention -> obar (bf16, [B*Nw, 512])
        k_window_attn<<<Nw, 256, 0, stream>>>(QKV, obar, w, Nw);

        // Wo[s] -> bf16
        {
            const int n = D * D;
            k_f32_to_bf16<<<(n + 255) / 256, 256, 0, stream>>>(
                Wo + (size_t)s * n, Woh, n);
        }

        // Y = obar @ Wo^T + bo
        {
            const int M = B * Nw, N = D, K = D;
            const int tiles = ((M + 31) / 32) * (N / 64);
            k_gemm_bf16<<<(tiles + 7) / 8, 256, 0, stream>>>(
                obar, Woh, bo + (size_t)s * D, Y, M, N, K, /*qcols=*/0, 1.0f);
        }

        // accumulate 0.25 * interp(Y) into out
        {
            const int n = B * T * D;
            k_interp_acc<<<(n + 255) / 256, 256, 0, stream>>>(Y, out, Nw, s == 0);
        }
    }
}
